// HDMetric_14826227105999
// MI455X (gfx1250) — compile-verified, matrix operand order checked by reading, not run on hardware
//
#include <hip/hip_runtime.h>
#include <hip/hip_bf16.h>
#include <math.h>

#define Bx 8
#define NC 4
#define NFG 3
#define H 192
#define W 192
#define HW (H*W)
#define NPAIR (Bx*NFG)      // 24 (b, class) pairs
#define NMAP  (NPAIR*2)     // 48 directed distance maps
#define NB 73984            // squared-dist bins 0..72962 + sentinel (NB = 256*289)
#define SENTBIN (NB-1)
#define BIGU 1000000000u    // matches reference BIG=1e9 penalty

typedef __attribute__((ext_vector_type(8))) int v8i;

// ---- workspace layout (bytes) ----
#define O_PRED   ((size_t)0)
#define O_SURFP  (O_PRED  + (size_t)Bx*HW)          // 294912, 32B aligned
#define O_SURFG  (O_SURFP + (size_t)NPAIR*HW)       // +884736
#define O_G      (O_SURFG + (size_t)NPAIR*HW)       // u32 g maps, 48*HW*4
#define O_HIST   (O_G     + (size_t)NMAP*HW*4)      // u32 hist, 48*NB*4
#define O_N      (O_HIST  + (size_t)NMAP*NB*4)      // u32 counts[48]
#define O_Q      (O_N     + (size_t)256)            // float q100[48], q95[48]

// ---------- 1) argmax over channels (softmax is monotonic -> skip it) ----------
__global__ void k_argmax(const float* __restrict__ logits, unsigned char* __restrict__ pred) {
    int i = blockIdx.x * blockDim.x + threadIdx.x;
    if (i >= Bx * HW) return;
    int b = i / HW, px = i % HW;
    const float* base = logits + (size_t)b * NC * HW + px;
    float best = base[0]; int bi = 0;
    #pragma unroll
    for (int c = 1; c < NC; ++c) {
        float v = base[(size_t)c * HW];
        if (v > best) { best = v; bi = c; }   // strict > keeps first max (jnp.argmax)
    }
    pred[i] = (unsigned char)bi;
}

// ---------- 2) 4-neighbor surface extraction (border counts as background) ----------
__global__ void k_surface(const unsigned char* __restrict__ pred, const int* __restrict__ tgt,
                          unsigned char* __restrict__ sp, unsigned char* __restrict__ sg) {
    int i = blockIdx.x * blockDim.x + threadIdx.x;
    if (i >= NPAIR * HW) return;
    int p = i / HW, px = i % HW;
    int b = p / NFG, cls = (p % NFG) + 1;
    int h = px / W, w = px % W;
    const unsigned char* pb = pred + (size_t)b * HW;
    const int* tb = tgt + (size_t)b * HW;

    bool fg = pb[px] == (unsigned char)cls;
    bool up = (h > 0)     && pb[px - W] == (unsigned char)cls;
    bool dn = (h < H - 1) && pb[px + W] == (unsigned char)cls;
    bool lf = (w > 0)     && pb[px - 1] == (unsigned char)cls;
    bool rt = (w < W - 1) && pb[px + 1] == (unsigned char)cls;
    sp[i] = (fg && !(up && dn && lf && rt)) ? 1 : 0;

    bool gfg = tb[px] == cls;
    bool gup = (h > 0)     && tb[px - W] == cls;
    bool gdn = (h < H - 1) && tb[px + W] == cls;
    bool glf = (w > 0)     && tb[px - 1] == cls;
    bool grt = (w < W - 1) && tb[px + 1] == cls;
    sg[i] = (gfg && !(gup && gdn && glf && grt)) ? 1 : 0;
}

// ---------- 3) exact 1D EDT along rows (per column): nearest-source two-scan ----------
// g[r][c] = min_{r'} (r-r')^2 over source rows in this column, or BIGU if column empty.
__global__ void k_edt_rows(const unsigned char* __restrict__ surfP,
                           const unsigned char* __restrict__ surfG,
                           unsigned* __restrict__ g) {
    int i = blockIdx.x * blockDim.x + threadIdx.x;
    if (i >= NMAP * W) return;
    int m = i / W, c = i % W;
    int p = m >> 1, s = m & 1;                    // s=0: sources = pred surface
    const unsigned char* src = (s == 0 ? surfP : surfG) + (size_t)p * HW;
    unsigned* gm = g + (size_t)m * HW;
    int last = -1000000;
    for (int r = 0; r < H; ++r) {                 // coalesced across threads at each r
        if (src[r * W + c]) last = r;
        gm[r * W + c] = (unsigned)(r - last);
    }
    int nxt = 1000000;
    for (int r = H - 1; r >= 0; --r) {
        if (src[r * W + c]) nxt = r;
        unsigned df = gm[r * W + c];
        unsigned db = (unsigned)(nxt - r);
        unsigned d = df < db ? df : db;
        gm[r * W + c] = (d < 100000u) ? d * d : BIGU;
    }
}

// ---------- 4) min-plus column pass from LDS + histogram of squared distances ----------
__global__ void __launch_bounds__(W) k_colpass_hist(const unsigned* __restrict__ g,
        const unsigned char* __restrict__ surfP, const unsigned char* __restrict__ surfG,
        unsigned* __restrict__ hist) {
    __shared__ unsigned row[W];
    int blk = blockIdx.x;                 // = m*H + r
    int m = blk / H, r = blk % H;
    int p = m >> 1, s = m & 1;
    int c = threadIdx.x;
    row[c] = g[(size_t)m * HW + r * W + c];
    __syncthreads();
    unsigned fmin = 0xFFFFFFFFu;
    for (int cp = 0; cp < W; ++cp) {
        int dc = c - cp;
        unsigned v = row[cp] + (unsigned)(dc * dc);
        fmin = v < fmin ? v : fmin;
    }
    // evaluation mask: dist-to-pred-surface sampled at gt surface, and vice versa
    const unsigned char* mask = (s == 0 ? surfG : surfP) + (size_t)p * HW;
    if (mask[r * W + c]) {
        unsigned bin = fmin <= 72962u ? fmin : (unsigned)SENTBIN;
        atomicAdd(&hist[(size_t)m * NB + bin], 1u);
    }
}

// ---------- 5) surface-voxel counts via V_WMMA_I32_16X16X64_IU8 ----------
// n = sum of 0/1 mask bytes. With B = all-ones, each WMMA yields sum(D) = 16 * sum(A bytes);
// accumulate in the i32 WMMA accumulator over tiles, wave-reduce, divide by 16.
__global__ void __launch_bounds__(256) k_count_wmma(const unsigned char* __restrict__ surfP,
        const unsigned char* __restrict__ surfG, unsigned* __restrict__ nArr) {
    int m = blockIdx.x;
    int p = m >> 1, s = m & 1;
    const unsigned char* mask = (s == 0 ? surfG : surfP) + (size_t)p * HW;
    int lane = threadIdx.x & 31;
    int wave = threadIdx.x >> 5;
    v8i ones;
    #pragma unroll
    for (int k = 0; k < 8; ++k) ones[k] = 0x01010101;
    v8i acc = {0, 0, 0, 0, 0, 0, 0, 0};
    // 36 tiles of 1024 bytes (16x64 iu8 A-operand per wave), 8 waves round-robin
    for (int t = wave; t < HW / 1024; t += 8) {
        v8i a = *(const v8i*)(mask + t * 1024 + lane * 32);   // 32B-aligned
        acc = __builtin_amdgcn_wmma_i32_16x16x64_iu8(false, a, false, ones, acc, false, false);
    }
    int ssum = 0;
    #pragma unroll
    for (int k = 0; k < 8; ++k) ssum += acc[k];
    for (int off = 16; off > 0; off >>= 1) ssum += __shfl_xor(ssum, off, 32);
    if (lane == 0) atomicAdd(&nArr[m], (unsigned)(ssum >> 4));
}

// ---------- 6) k-th smallest from histogram: chunk sums + serial refine ----------
__global__ void __launch_bounds__(256) k_stat(const unsigned* __restrict__ hist,
        const unsigned* __restrict__ nArr, float* __restrict__ q100, float* __restrict__ q95) {
    __shared__ unsigned csum[256];
    int m = blockIdx.x;
    const unsigned* hm = hist + (size_t)m * NB;
    int t = threadIdx.x;
    const int CH = NB / 256;                       // 289 bins per thread, exact
    unsigned sloc = 0;
    for (int k = 0; k < CH; ++k) sloc += hm[t * CH + k];
    csum[t] = sloc;
    __syncthreads();
    if (t == 0) {
        unsigned nn = nArr[m];
        unsigned k95 = (unsigned)ceilf(0.95f * (float)nn);
        if (k95 < 1u) k95 = 1u;
        if (nn > 0u && k95 > nn) k95 = nn;
        unsigned ks[2] = { nn, k95 };              // k=n -> max -> hd100
        float q[2];
        for (int qi = 0; qi < 2; ++qi) {
            if (nn == 0u) { q[qi] = __builtin_inff(); continue; }
            unsigned k = ks[qi];
            unsigned cum = 0; int bin = SENTBIN;
            for (int ch = 0; ch < 256; ++ch) {
                if (cum + csum[ch] >= k) {
                    unsigned c2 = cum;
                    for (int kk = 0; kk < CH; ++kk) {
                        c2 += hm[ch * CH + kk];
                        if (c2 >= k) { bin = ch * CH + kk; break; }
                    }
                    break;
                }
                cum += csum[ch];
            }
            q[qi] = (bin >= 72963) ? sqrtf(1.0e9f) : sqrtf((float)bin);
        }
        q100[m] = q[0];
        q95[m]  = q[1];
    }
}

// ---------- 7) robust = max over directions; micro mean over 24 pairs ----------
__global__ void k_final(const float* __restrict__ q100, const float* __restrict__ q95,
                        float* __restrict__ out) {
    if (blockIdx.x == 0 && threadIdx.x == 0) {
        float s100 = 0.f, s95 = 0.f;
        for (int p = 0; p < NPAIR; ++p) {
            s100 += fmaxf(q100[2 * p], q100[2 * p + 1]);
            s95  += fmaxf(q95[2 * p],  q95[2 * p + 1]);
        }
        out[0] = s100 / (float)NPAIR;
        out[1] = s95  / (float)NPAIR;
    }
}

extern "C" void kernel_launch(void* const* d_in, const int* in_sizes, int n_in,
                              void* d_out, int out_size, void* d_ws, size_t ws_size,
                              hipStream_t stream) {
    (void)in_sizes; (void)n_in; (void)out_size; (void)ws_size;
    const float* logits = (const float*)d_in[0];
    const int*   target = (const int*)d_in[1];
    float* out = (float*)d_out;
    char* ws = (char*)d_ws;

    unsigned char* pred = (unsigned char*)(ws + O_PRED);
    unsigned char* sp   = (unsigned char*)(ws + O_SURFP);
    unsigned char* sg   = (unsigned char*)(ws + O_SURFG);
    unsigned*      g    = (unsigned*)(ws + O_G);
    unsigned*      hist = (unsigned*)(ws + O_HIST);
    unsigned*      nArr = (unsigned*)(ws + O_N);
    float*         q100 = (float*)(ws + O_Q);
    float*         q95  = q100 + NMAP;

    // zero histogram + counts every call (capture-safe async memset)
    hipMemsetAsync(hist, 0, (size_t)NMAP * NB * 4 + 256, stream);

    k_argmax      <<<(Bx * HW + 255) / 256,    256, 0, stream>>>(logits, pred);
    k_surface     <<<(NPAIR * HW + 255) / 256, 256, 0, stream>>>(pred, target, sp, sg);
    k_edt_rows    <<<(NMAP * W + 255) / 256,   256, 0, stream>>>(sp, sg, g);
    k_colpass_hist<<<NMAP * H,                 W,   0, stream>>>(g, sp, sg, hist);
    k_count_wmma  <<<NMAP,                     256, 0, stream>>>(sp, sg, nArr);
    k_stat        <<<NMAP,                     256, 0, stream>>>(hist, nArr, q100, q95);
    k_final       <<<1,                        32,  0, stream>>>(q100, q95, out);
}
